// WordEntitySelfAttention_8787503087831
// MI455X (gfx1250) — matrix operand order repaired and checked by reading, not compiled, not run on hardware
//
#include <hip/hip_runtime.h>
#include <hip/hip_bf16.h>

typedef _Float16 f16;
typedef __attribute__((ext_vector_type(16))) _Float16 v16h;
typedef __attribute__((ext_vector_type(8)))  _Float16 v8h;
typedef __attribute__((ext_vector_type(4)))  _Float16 v4h;
typedef __attribute__((ext_vector_type(8)))  float    v8f;
typedef __attribute__((ext_vector_type(4))) unsigned int u32x4;
typedef __attribute__((ext_vector_type(8))) int          i32x8;
typedef __attribute__((ext_vector_type(4))) int          i32x4;

#define BB 8
#define LW 512
#define LE 128
#define DD 1024
#define HH 16
#define DH 64
#define SS 640

// ---------------- workspace layout (units: f16 halves) ----------------
static constexpr size_t MW      = 1024ull * 1024ull;
static constexpr size_t OFF_XW  = 0;               // 4 MW  (B*LW x D)
static constexpr size_t OFF_XE  = OFF_XW  + 4*MW;  // 1 MW  (B*LE x D)
static constexpr size_t OFF_WTQ = OFF_XE  + 1*MW;  // 4 MW  (4 x [N][K])
static constexpr size_t OFF_WTK = OFF_WTQ + 4*MW;  // 4 MW
static constexpr size_t OFF_WTV = OFF_WTK + 4*MW;  // 4 MW
static constexpr size_t OFF_QW0 = OFF_WTV + 4*MW;  // 4 MW  [b][h][512][64]
static constexpr size_t OFF_QW1 = OFF_QW0 + 4*MW;  // 4 MW
static constexpr size_t OFF_QE2 = OFF_QW1 + 4*MW;  // 1 MW  [b][h][128][64]
static constexpr size_t OFF_QE3 = OFF_QE2 + 1*MW;  // 1 MW
static constexpr size_t OFF_KW0 = OFF_QE3 + 1*MW;  // 4 MW  [b][h][512][64]
static constexpr size_t OFF_KW2 = OFF_KW0 + 4*MW;  // 4 MW
static constexpr size_t OFF_KE1 = OFF_KW2 + 4*MW;  // 1 MW  [b][h][128][64]
static constexpr size_t OFF_KE3 = OFF_KE1 + 1*MW;  // 1 MW
static constexpr size_t OFF_VT0 = OFF_KE3 + 1*MW;  // 4 MW  [b][h][64][512]
static constexpr size_t OFF_VT2 = OFF_VT0 + 4*MW;  // 4 MW
static constexpr size_t OFF_VT1 = OFF_VT2 + 4*MW;  // 1 MW  [b][h][64][128]
static constexpr size_t OFF_VT3 = OFF_VT1 + 1*MW;  // 1 MW

// ---------------- WMMA helpers (gfx1250 layouts, ISA 7.12.2) ----------------
__device__ __forceinline__ v8f wmma_f16(v16h a, v16h b, v8f c) {
  return __builtin_amdgcn_wmma_f32_16x16x32_f16(
      /*neg_a=*/false, a, /*neg_b=*/false, b,
      /*c_mod=*/(short)0, c, /*reuse_a=*/false, /*reuse_b=*/false);
}

// A tile 16x32 (MxK) from row-major src (lda in halves). src points at (m0,k0).
__device__ __forceinline__ v16h load_a(const f16* __restrict__ src, int lda) {
  int lane = threadIdx.x & 31;
  int m  = lane & 15;
  int kb = (lane >> 4) * 16;
  const f16* p = src + (size_t)m * lda + kb;
  v8h lo = *(const v8h*)(p);        // K = kb .. kb+7
  v8h hi = *(const v8h*)(p + 8);    // K = kb+8 .. kb+15
  v16h a;
#pragma unroll
  for (int i = 0; i < 8; ++i) { a[i] = lo[i]; a[i + 8] = hi[i]; }
  return a;
}

// B tile 32x16 (KxN) from "N-major" layout bt[n][k] (ldb in halves). bt -> (n0,k0).
__device__ __forceinline__ v16h load_b(const f16* __restrict__ bt, int ldb) {
  int lane = threadIdx.x & 31;
  int n  = lane & 15;
  int kb = (lane >> 4) * 16;
  return *(const v16h*)(bt + (size_t)n * ldb + kb);   // 16 contiguous halves
}

// B tile 32x16 from LDS, rows of 32 halves packed (ld = 32)
__device__ __forceinline__ v16h load_b_lds32(const f16* lds) {
  int lane = threadIdx.x & 31;
  int n  = lane & 15;
  int kb = (lane >> 4) * 16;
  const f16* p = lds + n * 32 + kb;
  v8h lo = *(const v8h*)(p);
  v8h hi = *(const v8h*)(p + 8);
  v16h b;
#pragma unroll
  for (int i = 0; i < 8; ++i) { b[i] = lo[i]; b[i + 8] = hi[i]; }
  return b;
}

// A tile 16x32 from LDS row-major [16][32]
__device__ __forceinline__ v16h load_a_lds(const f16* lds) {
  int lane = threadIdx.x & 31;
  int m  = lane & 15;
  int kb = (lane >> 4) * 16;
  v16h a;
#pragma unroll
  for (int i = 0; i < 8; ++i) {
    a[i]     = lds[m * 32 + kb + i];
    a[i + 8] = lds[m * 32 + kb + 8 + i];
  }
  return a;
}

__device__ __forceinline__ float half_rmax(float v) {
  v = fmaxf(v, __shfl_xor(v, 1, 32));
  v = fmaxf(v, __shfl_xor(v, 2, 32));
  v = fmaxf(v, __shfl_xor(v, 4, 32));
  v = fmaxf(v, __shfl_xor(v, 8, 32));
  return v;
}
__device__ __forceinline__ float half_rsum(float v) {
  v += __shfl_xor(v, 1, 32);
  v += __shfl_xor(v, 2, 32);
  v += __shfl_xor(v, 4, 32);
  v += __shfl_xor(v, 8, 32);
  return v;
}

// prefetch 32 consecutive 64-half rows (one 128B line per lane) into cache
__device__ __forceinline__ void prefetch_rows(const f16* p) {
  int lane = threadIdx.x & 31;
  __builtin_prefetch(p + (size_t)lane * 64, 0, 3);
}

// ---------------- TDM: 2-D tile load Global -> LDS (ISA ch.8 D#) ----------
// Copies tile_d1 rows x tile_d0 elements (2-byte), row stride = stride elems,
// into LDS at lds_byte_off (rows packed contiguously).
__device__ __forceinline__ void tdm_load_2d(unsigned lds_byte_off, const void* gptr,
                                            unsigned tdim0, unsigned tdim1,
                                            unsigned tile_d0, unsigned tile_d1,
                                            unsigned stride) {
  unsigned long long ga = (unsigned long long)(size_t)gptr;
  u32x4 g0;
  g0[0] = 1u;                                             // count=1, user D#
  g0[1] = lds_byte_off;                                   // lds_addr (bytes)
  g0[2] = (unsigned)(ga & 0xffffffffu);                   // global_addr[31:0]
  g0[3] = (unsigned)((ga >> 32) & 0x01ffffffu) | (2u << 30); // addr[56:32] | type=2
  i32x8 g1;
  g1[0] = (int)(1u << 16);                                // data_size=1 (2 bytes)
  g1[1] = (int)((tdim0 & 0xffffu) << 16);                 // tensor_dim0[15:0]
  g1[2] = (int)(((tdim0 >> 16) & 0xffffu) | ((tdim1 & 0xffffu) << 16));
  g1[3] = (int)(((tdim1 >> 16) & 0xffffu) | ((tile_d0 & 0xffffu) << 16));
  g1[4] = (int)(tile_d1 & 0xffffu);                       // tile_dim1 (tile_dim2=0)
  g1[5] = (int)stride;                                    // tensor_dim0_stride[31:0]
  g1[6] = 0;                                              // stride[47:32], dim1_stride
  g1[7] = 0;
  i32x4 z4 = {0, 0, 0, 0};
  i32x8 z8 = {0, 0, 0, 0, 0, 0, 0, 0};
  __builtin_amdgcn_tensor_load_to_lds(g0, g1, z4, z4, z8, 0);
}

// ---------------- kernel 1a: f32 -> f16 convert ----------------
__global__ void luke_cvt_f16(const float* __restrict__ in, f16* __restrict__ o, int n) {
  int i = (blockIdx.x * blockDim.x + threadIdx.x) * 4;
  if (i < n) {
    float4 v = *(const float4*)(in + i);
    v4h h; h[0] = (f16)v.x; h[1] = (f16)v.y; h[2] = (f16)v.z; h[3] = (f16)v.w;
    *(v4h*)(o + i) = h;
  }
}

// ---------------- kernel 1b: W[k][n] f32 -> Wt[n][k] f16 (tiled transpose) ----
__global__ void luke_wtrans(const float* __restrict__ w, f16* __restrict__ wt) {
  __shared__ float tile[32][33];
  const float* wp = w  + (size_t)blockIdx.z * MW;
  f16*        wtp = wt + (size_t)blockIdx.z * MW;
  int k0 = blockIdx.x * 32, n0 = blockIdx.y * 32;
  int tx = threadIdx.x, ty = threadIdx.y;           // 32 x 8
#pragma unroll
  for (int i = 0; i < 32; i += 8)
    tile[ty + i][tx] = wp[(size_t)(k0 + ty + i) * DD + (n0 + tx)];
  __syncthreads();
#pragma unroll
  for (int i = 0; i < 32; i += 8)
    wtp[(size_t)(n0 + ty + i) * DD + (k0 + tx)] = (f16)tile[tx][ty + i];
}

// ---------------- kernel 2: projection GEMM Y = X*W + b --------------------
// block = 4 waves -> 16 rows x 256 cols. Per K-step the B panel (256x32 f16,
// 16KB) is staged Global->LDS by the Tensor Data Mover (wave 0 issues, TENSORcnt
// waits), double buffered; WMMAs consume from LDS. A tile is tiny & L0-resident.
// out_mode 0: [b][h][s][64], 1: [b][h][64][s]
__global__ __launch_bounds__(128) void luke_proj(
    const f16* __restrict__ X, const f16* __restrict__ Wt,
    const float* __restrict__ bias, f16* __restrict__ out,
    int rows_pb, int out_mode) {
  __shared__ f16 bpan[2][256 * 32];                 // 2 x 16KB ping-pong
  int lane = threadIdx.x & 31;
  int w    = threadIdx.x >> 5;                      // wave id 0..3
  int rowtiles = rows_pb >> 4;
  int ngq = blockIdx.x & 3;                         // quad of col-groups
  int rt  = (blockIdx.x >> 2) % rowtiles;
  int b   = (blockIdx.x >> 2) / rowtiles;
  if (b >= BB) return;
  int ng  = ngq * 4 + w;                            // this wave's col-group of 64

  const f16* xrow = X + (size_t)(b * rows_pb + rt * 16) * DD;
  const f16* wpan = Wt + (size_t)(ngq * 256) * DD;  // 256-row B panel base

  // prologue: DMA B panel for kt=0 into buffer 0
  if (threadIdx.x < 32) {
    tdm_load_2d((unsigned)(size_t)&bpan[0][0], wpan,
                DD, 256, 32, 256, DD);
  }

  v8f acc[4] = {v8f{}, v8f{}, v8f{}, v8f{}};

  for (int kt = 0; kt < DD / 32; ++kt) {
    int p = kt & 1;
    if (threadIdx.x < 32) {
      if (kt < DD / 32 - 1) {
        // issue next panel into the other buffer, keep it in flight
        tdm_load_2d((unsigned)(size_t)&bpan[p ^ 1][0], wpan + (kt + 1) * 32,
                    DD, 256, 32, 256, DD);
        __builtin_amdgcn_s_wait_tensorcnt(1);       // wait for kt's DMA only
      } else {
        __builtin_amdgcn_s_wait_tensorcnt(0);
      }
    }
    __syncthreads();                                // LDS stage visible to all waves

    v16h a = load_a(xrow + kt * 32, DD);            // shared 16x32 A tile (L0 hit)
    const f16* bp = &bpan[p][0] + (w * 64) * 32;    // this wave's 64 rows
    acc[0] = wmma_f16(a, load_b_lds32(bp + 0 * 16 * 32), acc[0]);
    acc[1] = wmma_f16(a, load_b_lds32(bp + 1 * 16 * 32), acc[1]);
    acc[2] = wmma_f16(a, load_b_lds32(bp + 2 * 16 * 32), acc[2]);
    acc[3] = wmma_f16(a, load_b_lds32(bp + 3 * 16 * 32), acc[3]);

    __syncthreads();                                // buffer free for next DMA
  }

  int hi = lane >> 4, nl = lane & 15;
#pragma unroll
  for (int j = 0; j < 4; ++j) {
    int n  = ng * 64 + j * 16 + nl;
    float bv = bias[n];
    int h = n >> 6, dh = n & 63;
    if (out_mode == 0) {                              // [b][h][s][64]
      f16* op = out + (((size_t)b * HH + h) * rows_pb) * 64;
#pragma unroll
      for (int r = 0; r < 8; ++r) {
        int s = rt * 16 + 8 * hi + r;
        op[(size_t)s * 64 + dh] = (f16)(acc[j][r] + bv);
      }
    } else {                                          // [b][h][64][s]
      f16* op = out + (((size_t)b * HH + h) * 64 + dh) * rows_pb + rt * 16 + 8 * hi;
      v8h pk;
#pragma unroll
      for (int r = 0; r < 8; ++r) pk[r] = (f16)(acc[j][r] + bv);
      *(v8h*)op = pk;
    }
  }
}

// ---------------- kernel 3: flash attention, one wave per (b,h,16 queries) --
__device__ __forceinline__ void attn_chunk(
    const v16h& a_lo, const v16h& a_hi,
    const f16* __restrict__ Kmat, const f16* __restrict__ Vt, int kvlen,
    int k0, const float* __restrict__ maskp,
    float* mrun, float* lrun, v8f* acc, f16* lds) {
  int lane = threadIdx.x & 31;

  // issue ALL tile loads up front: 4 K-tiles + 4 V-tiles (V independent of softmax)
  v16h kb0 = load_b(Kmat + (size_t)k0 * 64, 64);
  v16h kb1 = load_b(Kmat + (size_t)k0 * 64 + 32, 64);
  v16h kb2 = load_b(Kmat + (size_t)(k0 + 16) * 64, 64);
  v16h kb3 = load_b(Kmat + (size_t)(k0 + 16) * 64 + 32, 64);
  v16h vb0 = load_b(Vt + (size_t)(0 * 16) * kvlen + k0, kvlen);
  v16h vb1 = load_b(Vt + (size_t)(1 * 16) * kvlen + k0, kvlen);
  v16h vb2 = load_b(Vt + (size_t)(2 * 16) * kvlen + k0, kvlen);
  v16h vb3 = load_b(Vt + (size_t)(3 * 16) * kvlen + k0, kvlen);

  // scores: two 16x16 tiles over keys [k0,k0+16), [k0+16,k0+32)
  v8f s0 = {}, s1 = {};
  s0 = wmma_f16(a_lo, kb0, s0);
  s0 = wmma_f16(a_hi, kb1, s0);
  s1 = wmma_f16(a_lo, kb2, s1);
  s1 = wmma_f16(a_hi, kb3, s1);

  float mk0 = maskp[k0 + (lane & 15)];
  float mk1 = maskp[k0 + 16 + (lane & 15)];
  float p0[8], p1[8], alpha[8];
#pragma unroll
  for (int r = 0; r < 8; ++r) {
    float x0 = s0[r] * 0.125f + mk0;                  // 1/sqrt(64)
    float x1 = s1[r] * 0.125f + mk1;
    float cm = half_rmax(fmaxf(x0, x1));
    float mn = fmaxf(mrun[r], cm);
    alpha[r] = __expf(mrun[r] - mn);
    p0[r] = __expf(x0 - mn);
    p1[r] = __expf(x1 - mn);
    float rs = half_rsum(p0[r] + p1[r]);
    lrun[r] = lrun[r] * alpha[r] + rs;
    mrun[r] = mn;
  }
#pragma unroll
  for (int t = 0; t < 4; ++t)
#pragma unroll
    for (int r = 0; r < 8; ++r) acc[t][r] *= alpha[r];

  // stage P (16x32 f16) through LDS to reshape C-layout -> A-layout
  int hi = lane >> 4, nl = lane & 15;
#pragma unroll
  for (int r = 0; r < 8; ++r) {
    lds[(8 * hi + r) * 32 + nl]      = (f16)p0[r];
    lds[(8 * hi + r) * 32 + 16 + nl] = (f16)p1[r];
  }
  __syncthreads();
  v16h pa = load_a_lds(lds);
  acc[0] = wmma_f16(pa, vb0, acc[0]);
  acc[1] = wmma_f16(pa, vb1, acc[1]);
  acc[2] = wmma_f16(pa, vb2, acc[2]);
  acc[3] = wmma_f16(pa, vb3, acc[3]);
  __syncthreads();
}

__global__ __launch_bounds__(64) void luke_attn(
    const f16* __restrict__ ws, const float* __restrict__ mask,
    float* __restrict__ out) {
  __shared__ f16 plds[2][16 * 32];
  int wwarp = threadIdx.x >> 5;
  int lane  = threadIdx.x & 31;
  int wid = blockIdx.x * 2 + wwarp;
  int qt = wid % 40;                 // 40 query tiles of 16 over S=640
  int h  = (wid / 40) & 15;
  int b  = wid / 640;
  if (b >= BB) return;
  int s0 = qt * 16;
  bool isword = (s0 < LW);
  size_t bh = (size_t)b * HH + h;

  const f16 *Qw_, *Qe_, *Kw_, *Ke_, *Vw_, *Ve_;
  int sl;
  if (isword) {
    sl  = s0;
    Qw_ = ws + OFF_QW0 + bh * (size_t)LW * 64;
    Qe_ = ws + OFF_QW1 + bh * (size_t)LW * 64;
    Kw_ = ws + OFF_KW0 + bh * (size_t)LW * 64;
    Ke_ = ws + OFF_KE1 + bh * (size_t)LE * 64;
    Vw_ = ws + OFF_VT0 + bh * (size_t)64 * LW;
    Ve_ = ws + OFF_VT1 + bh * (size_t)64 * LE;
  } else {
    sl  = s0 - LW;
    Qw_ = ws + OFF_QE2 + bh * (size_t)LE * 64;
    Qe_ = ws + OFF_QE3 + bh * (size_t)LE * 64;
    Kw_ = ws + OFF_KW2 + bh * (size_t)LW * 64;
    Ke_ = ws + OFF_KE3 + bh * (size_t)LE * 64;
    Vw_ = ws + OFF_VT2 + bh * (size_t)64 * LW;
    Ve_ = ws + OFF_VT3 + bh * (size_t)64 * LE;
  }
  // Q rows resident in registers: 16x64 per projection flavor
  v16h aw_lo = load_a(Qw_ + (size_t)sl * 64, 64);
  v16h aw_hi = load_a(Qw_ + (size_t)sl * 64 + 32, 64);
  v16h ae_lo = load_a(Qe_ + (size_t)sl * 64, 64);
  v16h ae_hi = load_a(Qe_ + (size_t)sl * 64 + 32, 64);

  float mrun[8], lrun[8];
  v8f acc[4] = {v8f{}, v8f{}, v8f{}, v8f{}};
#pragma unroll
  for (int r = 0; r < 8; ++r) { mrun[r] = -1.0e30f; lrun[r] = 0.0f; }

  const float* mk = mask + (size_t)b * SS;
  f16* lds = plds[wwarp];

  for (int c = 0; c < LW / 32; ++c) {            // 16 word-key chunks
    if (c + 1 < LW / 32) prefetch_rows(Kw_ + (size_t)(c + 1) * 32 * 64);
    attn_chunk(aw_lo, aw_hi, Kw_, Vw_, LW, c * 32, mk, mrun, lrun, acc, lds);
  }
  for (int c = 0; c < LE / 32; ++c) {            // 4 entity-key chunks
    if (c + 1 < LE / 32) prefetch_rows(Ke_ + (size_t)(c + 1) * 32 * 64);
    attn_chunk(ae_lo, ae_hi, Ke_, Ve_, LE, c * 32, mk + LW, mrun, lrun, acc, lds);
  }

  int hi = lane >> 4, nl = lane & 15;
#pragma unroll
  for (int t = 0; t < 4; ++t) {
    int d = h * 64 + t * 16 + nl;
#pragma unroll
    for (int r = 0; r < 8; ++r) {
      int s = s0 + 8 * hi + r;
      float v = acc[t][r] * (1.0f / lrun[r]);
      size_t idx;
      if (s < LW) idx = ((size_t)b * LW + s) * DD + d;
      else        idx = (size_t)BB * LW * DD + ((size_t)b * LE + (s - LW)) * DD + d;
      out[idx] = v;
    }
  }
}

// ---------------- host launch ----------------
extern "C" void kernel_launch(void* const* d_in, const int* in_sizes, int n_in,
                              void* d_out, int out_size, void* d_ws, size_t ws_size,
                              hipStream_t stream) {
  const float* words = (const float*)d_in[0];
  const float* ents  = (const float*)d_in[1];
  const float* mask  = (const float*)d_in[2];
  const float* qw    = (const float*)d_in[3];
  const float* qb    = (const float*)d_in[4];
  const float* kw    = (const float*)d_in[5];
  const float* kb    = (const float*)d_in[6];
  const float* vw    = (const float*)d_in[7];
  const float* vb    = (const float*)d_in[8];
  float* out = (float*)d_out;
  f16* ws = (f16*)d_ws;

  // 1) convert hidden states to f16
  luke_cvt_f16<<<(BB * LW * DD / 4 + 255) / 256, 256, 0, stream>>>(words, ws + OFF_XW, BB * LW * DD);
  luke_cvt_f16<<<(BB * LE * DD / 4 + 255) / 256, 256, 0, stream>>>(ents,  ws + OFF_XE, BB * LE * DD);

  // 1b) transpose-convert weights: W[k][n] f32 -> Wt[n][k] f16
  dim3 tg(DD / 32, DD / 32, 4), tb(32, 8);
  luke_wtrans<<<tg, tb, 0, stream>>>(qw, ws + OFF_WTQ);
  luke_wtrans<<<tg, tb, 0, stream>>>(kw, ws + OFF_WTK);
  luke_wtrans<<<tg, tb, 0, stream>>>(vw, ws + OFF_WTV);

  // 2) 12 projection GEMMs: blocks = B * rowtiles * 4 (4 waves/block)
  int blkW = BB * (LW / 16) * 4;          // 1024 blocks, word-source jobs
  int blkE = BB * (LE / 16) * 4;          // 256 blocks, entity-source jobs
  // queries
  luke_proj<<<blkW, 128, 0, stream>>>(ws + OFF_XW, ws + OFF_WTQ + 0 * MW, qb + 0 * DD, ws + OFF_QW0, LW, 0);
  luke_proj<<<blkW, 128, 0, stream>>>(ws + OFF_XW, ws + OFF_WTQ + 1 * MW, qb + 1 * DD, ws + OFF_QW1, LW, 0);
  luke_proj<<<blkE, 128, 0, stream>>>(ws + OFF_XE, ws + OFF_WTQ + 2 * MW, qb + 2 * DD, ws + OFF_QE2, LE, 0);
  luke_proj<<<blkE, 128, 0, stream>>>(ws + OFF_XE, ws + OFF_WTQ + 3 * MW, qb + 3 * DD, ws + OFF_QE3, LE, 0);
  // keys
  luke_proj<<<blkW, 128, 0, stream>>>(ws + OFF_XW, ws + OFF_WTK + 0 * MW, kb + 0 * DD, ws + OFF_KW0, LW, 0);
  luke_proj<<<blkW, 128, 0, stream>>>(ws + OFF_XW, ws + OFF_WTK + 2 * MW, kb + 2 * DD, ws + OFF_KW2, LW, 0);
  luke_proj<<<blkE, 128, 0, stream>>>(ws + OFF_XE, ws + OFF_WTK + 1 * MW, kb + 1 * DD, ws + OFF_KE1, LE, 0);
  luke_proj<<<blkE, 128, 0, stream>>>(ws + OFF_XE, ws + OFF_WTK + 3 * MW, kb + 3 * DD, ws + OFF_KE3, LE, 0);
  // values (stored transposed [b][h][64][s])
  luke_proj<<<blkW, 128, 0, stream>>>(ws + OFF_XW, ws + OFF_WTV + 0 * MW, vb + 0 * DD, ws + OFF_VT0, LW, 1);
  luke_proj<<<blkW, 128, 0, stream>>>(ws + OFF_XW, ws + OFF_WTV + 2 * MW, vb + 2 * DD, ws + OFF_VT2, LW, 1);
  luke_proj<<<blkE, 128, 0, stream>>>(ws + OFF_XE, ws + OFF_WTV + 1 * MW, vb + 1 * DD, ws + OFF_VT1, LE, 1);
  luke_proj<<<blkE, 128, 0, stream>>>(ws + OFF_XE, ws + OFF_WTV + 3 * MW, vb + 3 * DD, ws + OFF_VT3, LE, 1);

  // 3) flash attention: 5120 waves, 2 waves/block
  luke_attn<<<(BB * HH * (SS / 16)) / 2, 64, 0, stream>>>(ws, mask, out);
}
